// Network_3272765080132
// MI455X (gfx1250) — matrix operand order repaired and checked by reading, not compiled, not run on hardware
//
#include <hip/hip_runtime.h>
#include <hip/hip_bf16.h>

typedef __attribute__((ext_vector_type(16))) __bf16 v16bf;
typedef __attribute__((ext_vector_type(8)))  float  v8f;

#define N_NODES 128
#define BVROWS  2048     // B * V
#define MT      256      // rows per feature-kernel block

// ---------------------------------------------------------------------------
// Kernel 0a: zero the f_blend accumulator (must be re-zeroed every call)
// ---------------------------------------------------------------------------
__global__ void k_zero(float* __restrict__ p, int n) {
  int i = blockIdx.x * blockDim.x + threadIdx.x;
  if (i < n) p[i] = 0.f;
}

// ---------------------------------------------------------------------------
// Kernel 0b: per-(b,v) inverse blend-weight sums: 1 / sum_n (bw[n,b,v]+1e-7)
// ---------------------------------------------------------------------------
__global__ void k_wsum(const float* __restrict__ bw, float* __restrict__ invs) {
  int r = blockIdx.x * blockDim.x + threadIdx.x;
  if (r >= BVROWS) return;
  float s = 0.f;
  for (int n = 0; n < N_NODES; ++n) s += bw[n * BVROWS + r] + 1e-7f;
  invs[r] = 1.f / s;
}

// ---------------------------------------------------------------------------
// Kernel 1: encode + decode (tiny: 128 nodes x B=2 rows). One block per node.
// d_out layout (floats): c[0..6143] d[6144..8191] ei[8192..16383]
//                        delta[16384..17151] mean[17152..19199] std[19200..21247]
// ---------------------------------------------------------------------------
__global__ void __launch_bounds__(64) k_encdec(
    const float* __restrict__ t_ped,   // [2,16]
    const float* __restrict__ pose,    // [2,72]
    const float* __restrict__ ec1_w, const float* __restrict__ ec1_b,
    const float* __restrict__ ec21_w, const float* __restrict__ ec21_b,
    const float* __restrict__ ec22_w, const float* __restrict__ ec22_b,
    const float* __restrict__ dc1_w, const float* __restrict__ dc1_b,
    const float* __restrict__ dc21_w, const float* __restrict__ dc21_b,
    const float* __restrict__ dc22_w, const float* __restrict__ dc22_b,
    float* __restrict__ out) {
  __shared__ float sIn[88];
  __shared__ float sNet[64];
  __shared__ float sMean[8];
  __shared__ float sDec[80];
  __shared__ float sDnet[64];
  const int n = blockIdx.x;
  const int t = threadIdx.x;

  for (int b = 0; b < 2; ++b) {
    if (t < 16) sIn[t] = t_ped[b * 16 + t];
    sIn[16 + t] = pose[b * 72 + t];            // t = 0..63
    if (t < 8) sIn[80 + t] = pose[b * 72 + 64 + t];
    __syncthreads();

    {
      float acc = ec1_b[n * 64 + t];
      for (int i = 0; i < 88; ++i) acc += sIn[i] * ec1_w[(n * 88 + i) * 64 + t];
      sNet[t] = acc > 0.f ? acc : 0.f;
    }
    __syncthreads();

    if (t < 8) {
      float m = ec21_b[n * 8 + t];
      float s = ec22_b[n * 8 + t];
      for (int i = 0; i < 64; ++i) {
        m += sNet[i] * ec21_w[(n * 64 + i) * 8 + t];
        s += sNet[i] * ec22_w[(n * 64 + i) * 8 + t];
      }
      sMean[t] = m;
      out[17152 + n * 16 + b * 8 + t] = m;                      // mean
      out[19200 + n * 16 + b * 8 + t] = 1.f / (1.f + expf(-s)); // std
    }
    __syncthreads();

    if (t < 8)  sDec[t] = sMean[t];
    sDec[8 + t] = sIn[16 + t];
    if (t < 8)  sDec[72 + t] = sIn[80 + t];
    __syncthreads();

    {
      float acc = dc1_b[n * 64 + t];
      for (int i = 0; i < 80; ++i) acc += sDec[i] * dc1_w[(n * 80 + i) * 64 + t];
      sDnet[t] = acc > 0.f ? acc : 0.f;
    }
    __syncthreads();

    if (t < 32) {
      float acc = dc21_b[n * 32 + t];
      for (int i = 0; i < 64; ++i) acc += sDnet[i] * dc21_w[(n * 64 + i) * 32 + t];
      out[8192 + n * 64 + b * 32 + t] = acc;   // ei
    }
    if (t < 3) {
      float acc = dc22_b[n * 3 + t];
      for (int i = 0; i < 64; ++i) acc += sDnet[i] * dc22_w[(n * 64 + i) * 3 + t];
      out[16384 + n * 6 + b * 3 + t] = acc;    // nodes_delta
    }
    __syncthreads();
  }
}

// ---------------------------------------------------------------------------
// WMMA fragment loaders (V_WMMA_F32_16X16X32_BF16; A/B use contiguous K, not the ISA table order)
// A (16x32 bf16): lane = M, VGPR v -> K pair (2v+16*half), 16 contiguous K per lane
// B (32x16 bf16) stored N-major in LDS: lane = N, 16 contiguous K per lane
// C/D (16x16 f32): lane = N, VGPR v -> M = v + 8*half
// ---------------------------------------------------------------------------
__device__ __forceinline__ v16bf load_a_frag(const __bf16* __restrict__ rowp,
                                             int k0, int half) {
  v16bf a;
#pragma unroll
  for (int v = 0; v < 8; ++v) {
    int kb = k0 + 16 * half + 2 * v;
    a[2 * v]     = rowp[kb];
    a[2 * v + 1] = rowp[kb + 1];
  }
  return a;
}

__device__ __forceinline__ v16bf load_b_frag(const __bf16* __restrict__ p) {
  v16bf b;
#pragma unroll
  for (int e = 0; e < 16; ++e) b[e] = p[e];
  return b;
}

// One MLP layer: out[MT,N] = relu(in[MT,K] @ W[K,N] + bias) via WMMA.
// 4 n-tiles accumulated per A-fragment load (4 live C frags = 32 VGPRs),
// so A-fragment LDS reads drop 4x vs one-tile-at-a-time.
__device__ __forceinline__ void layer_gemm(
    const __bf16* __restrict__ in, int instride, int K,
    const __bf16* __restrict__ wt, int kstride,
    const float* __restrict__ bias, int N,
    __bf16* __restrict__ out, int ostride, int wave, int lane) {
  const int half = lane >> 4;
  const int lmod = lane & 15;
  for (int mt = wave; mt < MT / 16; mt += 8) {
    const __bf16* inrow = in + (mt * 16 + lmod) * instride;
    for (int ng = 0; ng < N; ng += 64) {           // group of 4 n-tiles
      v8f c[4];
#pragma unroll
      for (int q = 0; q < 4; ++q) {
        float bval = bias[ng + q * 16 + lmod];
#pragma unroll
        for (int v = 0; v < 8; ++v) c[q][v] = bval;
      }
      for (int kk = 0; kk < K; kk += 32) {
        v16bf a = load_a_frag(inrow, kk, half);
#pragma unroll
        for (int q = 0; q < 4; ++q) {
          v16bf bm = load_b_frag(wt + (ng + q * 16 + lmod) * kstride + kk + (half << 4));
          c[q] = __builtin_amdgcn_wmma_f32_16x16x32_bf16(false, a, false, bm,
                                                         (short)0, c[q], false, false);
        }
      }
#pragma unroll
      for (int q = 0; q < 4; ++q) {
#pragma unroll
        for (int v = 0; v < 8; ++v) {
          float x = c[q][v];
          x = x > 0.f ? x : 0.f;
          int row = (mt << 4) + v + (half << 3);
          out[row * ostride + ng + (q << 4) + lmod] = (__bf16)x;
        }
      }
    }
  }
}

// ---------------------------------------------------------------------------
// Kernel 2: feature-field 4-layer per-node MLP (16-GFLOP hot loop) + masked
// blend accumulated into f_blend[2048,256] with native f32 global atomics.
// grid = (8 row-tiles, 128 nodes), 256 threads = 8 wave32. ~143 KB LDS.
// ---------------------------------------------------------------------------
__global__ void __launch_bounds__(256) k_feature(
    const float* __restrict__ ei,     // [128][2][32]  (d_out + 8192)
    const float* __restrict__ lc,     // [128][2][1024][63]
    const int*   __restrict__ mask,   // [128][2][1024]
    const float* __restrict__ bw,     // [128][2][1024]
    const float* __restrict__ invs,   // [2048]
    const float* __restrict__ w1, const float* __restrict__ b1,
    const float* __restrict__ w2, const float* __restrict__ b2,
    const float* __restrict__ w3, const float* __restrict__ b3,
    const float* __restrict__ w4, const float* __restrict__ b4,
    float* __restrict__ fblend) {    // [2048][256]
  __shared__ __bf16 sW1t[64 * 96];   // W1^T [N=64][K=96] (K padded 95->96)
  __shared__ __bf16 sW2t[64 * 64];
  __shared__ __bf16 sW3t[64 * 64];
  __shared__ __bf16 sW4t[256 * 64];
  __shared__ float  sB1[64], sB2[64], sB3[64], sB4[256];
  __shared__ float  sBw[MT];
  __shared__ float  sEi[32];
  __shared__ __bf16 sActA[MT * 96];  // X tile, later reused as L2 output
  __shared__ __bf16 sActB[MT * 64];  // L1 / L3 output

  const int tid  = threadIdx.x;
  const int n    = blockIdx.y;
  const int tile = blockIdx.x;
  const int b    = tile >> 2;
  const int r0   = tile * MT;

  // ---- weights -> bf16, transposed to N-major, into LDS ----
  const float* w1n = w1 + n * 95 * 64;
  for (int i = tid; i < 95 * 64; i += 256) {
    int k = i >> 6, c = i & 63;
    sW1t[c * 96 + k] = (__bf16)w1n[i];
  }
  if (tid < 64) sW1t[tid * 96 + 95] = (__bf16)0.f;
  const float* w2n = w2 + n * 64 * 64;
  const float* w3n = w3 + n * 64 * 64;
  for (int i = tid; i < 64 * 64; i += 256) {
    int k = i >> 6, c = i & 63;
    sW2t[c * 64 + k] = (__bf16)w2n[i];
    sW3t[c * 64 + k] = (__bf16)w3n[i];
  }
  const float* w4n = w4 + n * 64 * 256;
  for (int i = tid; i < 64 * 256; i += 256) {
    int k = i >> 8, c = i & 255;
    sW4t[c * 64 + k] = (__bf16)w4n[i];
  }
  if (tid < 64) {
    sB1[tid] = b1[n * 64 + tid];
    sB2[tid] = b2[n * 64 + tid];
    sB3[tid] = b3[n * 64 + tid];
  }
  sB4[tid] = b4[n * 256 + tid];
  if (tid < 32) sEi[tid] = ei[n * 64 + b * 32 + tid];
  __syncthreads();

  // ---- stage X tile [MT][96] = concat(ei[32], coords[63], pad) ----
  // coords for this tile are one contiguous 16128-float region -> coalesced
  {
    const int v0 = (tile & 3) * MT;
    const float* cbase = lc + ((size_t)(n * 2 + b) * 1024 + v0) * 63;
    for (int idx = tid; idx < MT * 63; idx += 256) {
      int row = idx / 63;
      int j   = idx - row * 63;
      sActA[row * 96 + 32 + j] = (__bf16)cbase[idx];
    }
    __bf16* xr = sActA + tid * 96;
#pragma unroll
    for (int j = 0; j < 32; ++j) xr[j] = (__bf16)sEi[j];
    xr[95] = (__bf16)0.f;
    int mi = (n * 2 + b) * 1024 + v0 + tid;
    float wv = bw[mi] + 1e-7f;                // mask folds into blend factor
    sBw[tid] = mask[mi] ? wv * invs[r0 + tid] : 0.f;
  }
  __syncthreads();

  const int wave = tid >> 5, lane = tid & 31;

  layer_gemm(sActA, 96, 96, sW1t, 96, sB1, 64, sActB, 64, wave, lane);
  __syncthreads();
  layer_gemm(sActB, 64, 64, sW2t, 64, sB2, 64, sActA, 64, wave, lane);
  __syncthreads();
  layer_gemm(sActA, 64, 64, sW3t, 64, sB3, 64, sActB, 64, wave, lane);
  __syncthreads();

  // ---- layer 4 (64->256) fused relu * blendweight + global f32 atomics ----
  {
    const int half = lane >> 4;
    const int lmod = lane & 15;
    for (int mt = wave; mt < MT / 16; mt += 8) {
      const __bf16* inrow = sActB + (mt * 16 + lmod) * 64;
      for (int ng = 0; ng < 256; ng += 64) {       // 4 n-tiles per A load
        v8f c[4];
#pragma unroll
        for (int q = 0; q < 4; ++q) {
          float bval = sB4[ng + q * 16 + lmod];
#pragma unroll
          for (int v = 0; v < 8; ++v) c[q][v] = bval;
        }
        for (int kk = 0; kk < 64; kk += 32) {
          v16bf a = load_a_frag(inrow, kk, half);
#pragma unroll
          for (int q = 0; q < 4; ++q) {
            v16bf bm = load_b_frag(sW4t + (ng + q * 16 + lmod) * 64 + kk + (half << 4));
            c[q] = __builtin_amdgcn_wmma_f32_16x16x32_bf16(false, a, false, bm,
                                                           (short)0, c[q], false, false);
          }
        }
#pragma unroll
        for (int q = 0; q < 4; ++q) {
#pragma unroll
          for (int v = 0; v < 8; ++v) {
            int row = (mt << 4) + v + (half << 3);
            float x = c[q][v];
            x = x > 0.f ? x : 0.f;
            x *= sBw[row];
            unsafeAtomicAdd(&fblend[(size_t)(r0 + row) * 256 + ng + (q << 4) + lmod], x);
          }
        }
      }
    }
  }
}

// ---------------------------------------------------------------------------
// Kernel 3: NeRF head (tiny: 34 MMAC). One block per (b,v) row.
// ---------------------------------------------------------------------------
__global__ void __launch_bounds__(64) k_head(
    const float* __restrict__ fblend,  // [2048][256]
    const float* __restrict__ c1w, const float* __restrict__ c1b,
    const float* __restrict__ c2w, const float* __restrict__ c2b,
    const float* __restrict__ d1w, const float* __restrict__ d1b,
    float* __restrict__ out) {
  __shared__ float sF[256];
  __shared__ float sNc[64];
  const int r = blockIdx.x;
  const int t = threadIdx.x;
  for (int i = t; i < 256; i += 64) sF[i] = fblend[r * 256 + i];
  __syncthreads();
  {
    float acc = c1b[t];
    for (int i = 0; i < 256; ++i) acc += sF[i] * c1w[i * 64 + t];
    sNc[t] = acc > 0.f ? acc : 0.f;
  }
  __syncthreads();
  if (t < 3) {
    float acc = c2b[t];
    for (int i = 0; i < 64; ++i) acc += sNc[i] * c2w[i * 3 + t];
    out[r * 3 + t] = acc;
  }
  if (t == 3) {
    float acc = d1b[0];
    for (int i = 0; i < 256; ++i) acc += sF[i] * d1w[i];
    out[6144 + r] = acc > 0.f ? acc : 0.f;
  }
}

// ---------------------------------------------------------------------------
extern "C" void kernel_launch(void* const* d_in, const int* in_sizes, int n_in,
                              void* d_out, int out_size, void* d_ws, size_t ws_size,
                              hipStream_t stream) {
  const float* t_ped  = (const float*)d_in[0];
  const float* pose   = (const float*)d_in[1];
  const float* lc     = (const float*)d_in[2];
  const int*   mask   = (const int*)d_in[3];
  const float* bwts   = (const float*)d_in[4];
  const float* ec1_w  = (const float*)d_in[5];
  const float* ec1_b  = (const float*)d_in[6];
  const float* ec21_w = (const float*)d_in[7];
  const float* ec21_b = (const float*)d_in[8];
  const float* ec22_w = (const float*)d_in[9];
  const float* ec22_b = (const float*)d_in[10];
  const float* dc1_w  = (const float*)d_in[11];
  const float* dc1_b  = (const float*)d_in[12];
  const float* dc21_w = (const float*)d_in[13];
  const float* dc21_b = (const float*)d_in[14];
  const float* dc22_w = (const float*)d_in[15];
  const float* dc22_b = (const float*)d_in[16];
  const float* f1_w   = (const float*)d_in[17];
  const float* f1_b   = (const float*)d_in[18];
  const float* f2_w   = (const float*)d_in[19];
  const float* f2_b   = (const float*)d_in[20];
  const float* f3_w   = (const float*)d_in[21];
  const float* f3_b   = (const float*)d_in[22];
  const float* f4_w   = (const float*)d_in[23];
  const float* f4_b   = (const float*)d_in[24];
  const float* c1_w   = (const float*)d_in[25];
  const float* c1_b   = (const float*)d_in[26];
  const float* c2_w   = (const float*)d_in[27];
  const float* c2_b   = (const float*)d_in[28];
  const float* d1_w   = (const float*)d_in[29];
  const float* d1_b   = (const float*)d_in[30];

  float* out    = (float*)d_out;
  float* fblend = (float*)d_ws;                 // 2048*256 f32 = 2 MB
  float* invs   = fblend + 2048 * 256;          // 2048 f32

  k_zero<<<(2048 * 256 + 255) / 256, 256, 0, stream>>>(fblend, 2048 * 256);
  k_wsum<<<8, 256, 0, stream>>>(bwts, invs);
  k_encdec<<<128, 64, 0, stream>>>(t_ped, pose, ec1_w, ec1_b, ec21_w, ec21_b,
                                   ec22_w, ec22_b, dc1_w, dc1_b, dc21_w, dc21_b,
                                   dc22_w, dc22_b, out);
  k_feature<<<dim3(8, 128), 256, 0, stream>>>(out + 8192, lc, mask, bwts, invs,
                                              f1_w, f1_b, f2_w, f2_b,
                                              f3_w, f3_b, f4_w, f4_b, fblend);
  k_head<<<2048, 64, 0, stream>>>(fblend, c1_w, c1_b, c2_w, c2_b, d1_w, d1_b, out);
}
